// CellWiseLSTM_86921548137267
// MI455X (gfx1250) — matrix operand order repaired and checked by reading, not compiled, not run on hardware
//
#include <hip/hip_runtime.h>

// ---------------- problem constants ----------------
#define SEQN  2048
#define BATCHN 64
#define HID   512
#define GATES 2048   // 4*HID
#define NBLK  16     // persistent grid size (must all be co-resident)

// ---------------- vector types ----------------
typedef __attribute__((ext_vector_type(16))) __bf16 v16bf;
typedef __attribute__((ext_vector_type(8)))  __bf16 v8bf;
typedef __attribute__((ext_vector_type(4)))  __bf16 v4bf;
typedef __attribute__((ext_vector_type(8)))  float  v8f;
typedef __attribute__((ext_vector_type(4)))  unsigned tdm_g0_t;   // D# group 0
typedef __attribute__((ext_vector_type(8)))  int      tdm_g1_t;   // D# group 1
typedef __attribute__((ext_vector_type(4)))  int      tdm_g23_t;  // D# groups 2/3
typedef __attribute__((ext_vector_type(8)))  int      tdm_g4_t;   // 6-arg variant extra

// ---------------- activation helpers ----------------
__device__ __forceinline__ float sigm(float x) {
  return 1.0f / (1.0f + __expf(-x));
}
__device__ __forceinline__ float tanh_fast(float x) {
  float e = __expf(-2.0f * fabsf(x));
  float r = (1.0f - e) / (1.0f + e);
  return x < 0.0f ? -r : r;
}

// A-operand (16x32 bf16, M x K) per-lane gather from LDS row (pitch 520):
// lanes 0-15: K k0..k0+15 contiguous
// lanes16-31: K k0+16..k0+31 contiguous
__device__ __forceinline__ v16bf load_a(const __bf16* row, int k0, int d) {
  union { v16bf v; v8bf h[2]; } u;
  u.h[0] = *(const v8bf*)(row + k0 + 2 * d);
  u.h[1] = *(const v8bf*)(row + k0 + 2 * d + 8);
  return u.v;
}

// ---------------- device-wide barrier (monotonic counter) ----------------
__device__ __forceinline__ void grid_barrier(unsigned* cnt, unsigned target) {
  __syncthreads();
  if (threadIdx.x == 0) {
    __hip_atomic_fetch_add(cnt, 1u, __ATOMIC_RELEASE, __HIP_MEMORY_SCOPE_AGENT);
    while (__hip_atomic_load(cnt, __ATOMIC_ACQUIRE, __HIP_MEMORY_SCOPE_AGENT) < target) {
      __builtin_amdgcn_s_sleep(8);
    }
  }
  __syncthreads();
}

// ---------------- Tensor Data Mover: contiguous 16KB global -> LDS, padded ----
// D# group0: count=1 | lds byte addr | global addr | type=2
__device__ __forceinline__ tdm_g0_t make_g0(unsigned lds_off, const void* gptr) {
  unsigned long long ga = (unsigned long long)(size_t)gptr;
  tdm_g0_t g0;
  g0.x = 1u;                                        // count=1, user mode
  g0.y = lds_off;                                   // LDS byte address
  g0.z = (unsigned)(ga & 0xFFFFFFFFu);              // global_addr[31:0]
  g0.w = (unsigned)((ga >> 32) & 0x01FFFFFFu)       // global_addr[56:32]
         | (2u << 30);                              // type = 2 ("image")
  return g0;
}
// D# group1: data_size=8B, pad_enable, pad_interval=256 DWORDs (1024B of data),
// pad_amount=4 DWORDs (16B) -> reproduces LDS pitch 1040B (= 520 bf16).
// tensor_dim0 = tile_dim0 = stride0 = 2048 units of 8B (one 16KB 1-D tile).
__device__ __forceinline__ tdm_g1_t make_g1() {
  tdm_g1_t g;
  g[0] = (3 << 16) | (1 << 20) | (7 << 22) | (3 << 25); // 0x07D30000
  g[1] = (int)(0x0800u << 16);   // tensor_dim0[15:0] in bits 63:48 (=2048)
  g[2] = (int)(1u << 16);        // tensor_dim0 hi=0 | tensor_dim1 lo16 = 1
  g[3] = (int)(0x0800u << 16);   // tensor_dim1 hi=0 | tile_dim0 = 2048
  g[4] = 0;                      // tile_dim1 = 0 (unused), tile_dim2 = 0
  g[5] = 2048;                   // tensor_dim0_stride low32
  g[6] = 0;
  g[7] = 0;
  return g;
}
__device__ __forceinline__ void tdm_load(tdm_g0_t g0, tdm_g1_t g1) {
  tdm_g23_t z = {0, 0, 0, 0};
#if __has_include(<hip/amd_detail/amd_gfx1250_TDM.h>)
  tdm_g4_t z8 = {0, 0, 0, 0, 0, 0, 0, 0};
  __builtin_amdgcn_tensor_load_to_lds(g0, g1, z, z, z8, 0);
#else
  __builtin_amdgcn_tensor_load_to_lds(g0, g1, z, z, 0);
#endif
}

// ---------------- LDS staging of x_t (fp32 -> bf16, pitch 520) ----------------
__device__ __forceinline__ void stage_x(__bf16* lds, const float* x, int t, int b0, int tid) {
  for (int idx = tid; idx < 16 * 128; idx += 256) {
    int row = idx >> 7;
    int c4  = (idx & 127) << 2;
    const float4 v = *(const float4*)(x + ((size_t)(b0 + row) * SEQN + t) * 512 + c4);
    v4bf pk = { (__bf16)v.x, (__bf16)v.y, (__bf16)v.z, (__bf16)v.w };
    *(v4bf*)(lds + row * 520 + c4) = pk;
  }
}

// ---------------- one LSTM cell tile: 4 gate tiles, K=512, 2 GEMMs ----------------
__device__ __forceinline__ void cell_tile(
    const __bf16* ldsX, const __bf16* ldsH,     // A operands, pitch 520
    const __bf16* Wih, const __bf16* Whh,       // [2048][512] bf16 row-major
    const float*  biasL,                        // [2048] combined b_ih+b_hh
    int j, int lane, float c[8], float hn[8])
{
  const int n  = lane & 15;
  const int d  = (lane >> 4) * 8;    // A chunk shift for lane half
  const int kb = (lane >> 4) * 16;   // B K shift for lane half
  const __bf16* arX = ldsX + (lane & 15) * 520;
  const __bf16* arH = ldsH + (lane & 15) * 520;

  v8f acc[4];
#pragma unroll
  for (int g = 0; g < 4; ++g) {
    float bv = biasL[g * HID + j * 16 + n];
    acc[g] = (v8f){bv, bv, bv, bv, bv, bv, bv, bv};
  }

#pragma unroll 4
  for (int kk = 0; kk < 16; ++kk) {
    const int k0 = kk * 32;
    v16bf aX = load_a(arX, k0, d);
    v16bf aH = load_a(arH, k0, d);
#pragma unroll
    for (int g = 0; g < 4; ++g) {
      // B operand: lane = gate column n, 16 consecutive K per lane half
      const size_t rb = (size_t)(g * HID + j * 16 + n) * 512 + (size_t)(k0 + kb);
      v16bf bX = *(const v16bf*)(Wih + rb);
      acc[g] = __builtin_amdgcn_wmma_f32_16x16x32_bf16(false, aX, false, bX,
                                                       (short)0, acc[g], false, false);
      v16bf bH = *(const v16bf*)(Whh + rb);
      acc[g] = __builtin_amdgcn_wmma_f32_16x16x32_bf16(false, aH, false, bH,
                                                       (short)0, acc[g], false, false);
    }
  }

#pragma unroll
  for (int r = 0; r < 8; ++r) {
    float iv = sigm(acc[0][r]);
    float fv = sigm(acc[1][r]);
    float gv = tanh_fast(acc[2][r]);
    float ov = sigm(acc[3][r]);
    float cn = fv * c[r] + iv * gv;
    c[r] = cn;
    hn[r] = ov * tanh_fast(cn);
  }
}

// ---------------- init: fp32->bf16 weights, combine bias, init h, reset barrier ----------------
__global__ __launch_bounds__(256) void lstm_init(
    const float* __restrict__ Wih, const float* __restrict__ Whh,
    const float* __restrict__ bih, const float* __restrict__ bhh,
    const float* __restrict__ hx,
    __bf16* __restrict__ Wih_bf, __bf16* __restrict__ Whh_bf,
    float* __restrict__ bias, __bf16* __restrict__ hbf,
    unsigned* __restrict__ cnt)
{
  const size_t tid  = (size_t)blockIdx.x * 256 + threadIdx.x;
  const size_t nthr = (size_t)gridDim.x * 256;
  const size_t NW = (size_t)2 * GATES * 512;
  for (size_t i = tid; i < NW; i += nthr) {
    Wih_bf[i] = (__bf16)Wih[i];
    Whh_bf[i] = (__bf16)Whh[i];
  }
  for (size_t i = tid; i < (size_t)2 * GATES; i += nthr) bias[i] = bih[i] + bhh[i];
  // hx -> phase 1 (read phase of t=0)
  for (size_t i = tid; i < (size_t)2 * BATCHN * HID; i += nthr) {
    size_t l = i / ((size_t)BATCHN * HID), r = i % ((size_t)BATCHN * HID);
    hbf[(l * 2 + 1) * (size_t)BATCHN * HID + r] = (__bf16)hx[i];
  }
  if (tid == 0) *cnt = 0u;
}

// ---------------- persistent recurrent kernel ----------------
__global__ __launch_bounds__(256, 1) void lstm_persist(
    const float* __restrict__ x, const float* __restrict__ cx,
    float* __restrict__ out,
    const __bf16* __restrict__ Wih, const __bf16* __restrict__ Whh,
    const float* __restrict__ bias, __bf16* __restrict__ hbf,
    unsigned* __restrict__ cnt)
{
  __shared__ __bf16 ldsX[16 * 520];
  __shared__ __bf16 ldsH[16 * 520];

  const int tid  = threadIdx.x;
  const int lane = tid & 31;
  const int wid  = tid >> 5;
  const int b0   = (blockIdx.x & 3) * 16;          // batch tile
  const int j    = (blockIdx.x >> 2) * 8 + wid;    // hidden col tile 0..31
  const int n    = lane & 15;
  const int hih  = lane >> 4;
  const int col  = j * 16 + n;

  // TDM descriptors (uniform per block): 16KB h-tile starting at row b0, per phase
  const unsigned ldsX_off = (unsigned)(size_t)(void*)&ldsX[0];
  const unsigned ldsH_off = (unsigned)(size_t)(void*)&ldsH[0];
  const tdm_g1_t G1 = make_g1();
  tdm_g0_t g0_L0H[2], g0_L1X[2], g0_L1H[2];
#pragma unroll
  for (int p = 0; p < 2; ++p) {
    const __bf16* h0p = hbf + (size_t)(0 * 2 + p) * BATCHN * HID + (size_t)b0 * HID;
    const __bf16* h1p = hbf + (size_t)(1 * 2 + p) * BATCHN * HID + (size_t)b0 * HID;
    g0_L0H[p] = make_g0(ldsH_off, h0p);
    g0_L1X[p] = make_g0(ldsX_off, h0p);
    g0_L1H[p] = make_g0(ldsH_off, h1p);
  }

  float c0[8], c1[8], h0r[8], h1r[8];
#pragma unroll
  for (int r = 0; r < 8; ++r) {
    int row = b0 + r + 8 * hih;
    c0[r] = cx[(size_t)(0 * BATCHN + row) * HID + col];
    c1[r] = cx[(size_t)(1 * BATCHN + row) * HID + col];
    h0r[r] = 0.0f; h1r[r] = 0.0f;
  }

  unsigned bars = 0;
  for (int t = 0; t < SEQN; ++t) {
    const int rp = (t + 1) & 1;   // phase written at step t-1
    const int wp = t & 1;         // phase written at step t

    // ---- layer 0: gates = x_t Wih0^T + h0 Whh0^T + b ----
    if (wid == 0) tdm_load(g0_L0H[rp], G1);          // h0(prev) -> ldsH via TDM
    stage_x(ldsX, x, t, b0, tid);                    // x_t -> ldsX (cvt to bf16)
    if (wid == 0) __builtin_amdgcn_s_wait_tensorcnt(0);
    __syncthreads();
    cell_tile(ldsX, ldsH, Wih, Whh, bias, j, lane, c0, h0r);
    {
      __bf16* hdst = hbf + (size_t)(0 * 2 + wp) * BATCHN * HID;
#pragma unroll
      for (int r = 0; r < 8; ++r) {
        int row = b0 + r + 8 * hih;
        hdst[(size_t)row * HID + col] = (__bf16)h0r[r];
      }
    }
    grid_barrier(cnt, ++bars * NBLK);

    // ---- layer 1: gates = h0_new Wih1^T + h1 Whh1^T + b ----
    if (wid == 0) {
      tdm_load(g0_L1X[wp], G1);                      // h0(new)  -> ldsX
      tdm_load(g0_L1H[rp], G1);                      // h1(prev) -> ldsH
      __builtin_amdgcn_s_wait_tensorcnt(0);
    }
    __syncthreads();
    cell_tile(ldsX, ldsH, Wih + (size_t)GATES * 512, Whh + (size_t)GATES * 512,
              bias + GATES, j, lane, c1, h1r);
    {
      __bf16* hdst = hbf + (size_t)(1 * 2 + wp) * BATCHN * HID;
      float*  op   = out + (size_t)t * HID;
#pragma unroll
      for (int r = 0; r < 8; ++r) {
        int row = b0 + r + 8 * hih;
        hdst[(size_t)row * HID + col] = (__bf16)h1r[r];
        op[(size_t)row * SEQN * HID + col] = h1r[r];   // output[b, t, :]
      }
    }
    grid_barrier(cnt, ++bars * NBLK);
  }

  // ---- finals: h_n [2,B,H] then c_n [2,B,H] appended after output ----
  const size_t HN0 = (size_t)BATCHN * SEQN * HID;
  const size_t CN0 = HN0 + (size_t)2 * BATCHN * HID;
#pragma unroll
  for (int r = 0; r < 8; ++r) {
    int row = b0 + r + 8 * hih;
    out[HN0 + (size_t)row * HID + col]            = h0r[r];
    out[HN0 + (size_t)(BATCHN + row) * HID + col] = h1r[r];
    out[CN0 + (size_t)row * HID + col]            = c0[r];
    out[CN0 + (size_t)(BATCHN + row) * HID + col] = c1[r];
  }
}

// ---------------- launch ----------------
extern "C" void kernel_launch(void* const* d_in, const int* in_sizes, int n_in,
                              void* d_out, int out_size, void* d_ws, size_t ws_size,
                              hipStream_t stream) {
  const float* input_ = (const float*)d_in[0];
  const float* hx     = (const float*)d_in[1];
  const float* cx     = (const float*)d_in[2];
  const float* W_ih   = (const float*)d_in[3];
  const float* W_hh   = (const float*)d_in[4];
  const float* b_ih   = (const float*)d_in[5];
  const float* b_hh   = (const float*)d_in[6];
  float* out = (float*)d_out;
  char*  ws  = (char*)d_ws;

  // workspace layout (all offsets 256B-aligned):
  //   [0, 4MB)        Wih bf16  [2][2048][512]
  //   [4MB, 8MB)      Whh bf16  [2][2048][512]
  //   [8MB, +16KB)    bias f32  [2][2048] (b_ih+b_hh)
  //   [.., +256KB)    h bf16    [2 layers][2 phases][64][512]
  //   [.., +64B)      barrier counter
  __bf16*  Wih_bf = (__bf16*)(ws);
  __bf16*  Whh_bf = (__bf16*)(ws + (size_t)4 * 1024 * 1024);
  float*   bias   = (float*)(ws + (size_t)8 * 1024 * 1024);
  __bf16*  hbf    = (__bf16*)(ws + (size_t)8 * 1024 * 1024 + 16384);
  unsigned* cnt   = (unsigned*)(ws + (size_t)8 * 1024 * 1024 + 16384 + 262144);

  lstm_init<<<2048, 256, 0, stream>>>(W_ih, W_hh, b_ih, b_hh, hx,
                                      Wih_bf, Whh_bf, bias, hbf, cnt);
  lstm_persist<<<NBLK, 256, 0, stream>>>(input_, cx, out, Wih_bf, Whh_bf,
                                         bias, hbf, cnt);
}
